// DGCNNPredHead_63453846831166
// MI455X (gfx1250) — matrix operand order repaired and checked by reading, not compiled, not run on hardware
//
#include <hip/hip_runtime.h>
#include <hip/hip_bf16.h>
#include <math.h>

typedef __attribute__((ext_vector_type(2)))  float    v2f;
typedef __attribute__((ext_vector_type(8)))  float    v8f;
typedef __attribute__((ext_vector_type(16))) _Float16 v16h;

#define BATCH 8
#define CH    64
#define NPTS  4096
#define KNN   20
#define NEG_SLOPE 0.2f

__device__ __forceinline__ float lrelu(float v) { return v >= 0.0f ? v : NEG_SLOPE * v; }

// ---------------------------------------------------------------------------
// CDNA5 async global->LDS copy (ASYNCcnt-tracked, no VGPR round trip).
// GVS mode: mem_addr = SGPR_U64 + VGPR_I32 + IOFFSET; vdst = LDS byte addr.
// ---------------------------------------------------------------------------
__device__ __forceinline__ void async_b128(unsigned lds, unsigned goff, const void* base) {
    asm volatile("global_load_async_to_lds_b128 %0, %1, %2"
                 :: "v"(lds), "v"(goff), "s"(base) : "memory");
}
__device__ __forceinline__ void async_wait0() {
    asm volatile("s_wait_asynccnt 0x0" ::: "memory");
}
// Contiguous global->LDS async copy, 16B chunks, 256-thread block.
__device__ __forceinline__ void async_copy_lin(void* ldsDst, const void* src,
                                               int bytes, int tid) {
    unsigned base = (unsigned)(uintptr_t)ldsDst;   // flat->LDS: addr[31:0]
    for (int i = tid; i < (bytes >> 4); i += 256)
        async_b128(base + (unsigned)(i << 4), (unsigned)(i << 4), src);
}

// ---------------------------------------------------------------------------
// Prep: fold BN into (scale,bias), convert weights to f16.
// ---------------------------------------------------------------------------
__global__ __launch_bounds__(256) void prep_params(
    const float* wk1f, const float* gk1, const float* bk1, const float* mk1, const float* vk1,
    const float* wk2f, const float* gk2, const float* bk2, const float* mk2, const float* vk2,
    const float* w1f,  const float* g1,  const float* b1,  const float* m1,  const float* v1,
    const float* w2f,  const float* g2,  const float* b2,  const float* m2,  const float* v2,
    const float* w3,   const float* b3,
    _Float16* wk1h, _Float16* wk2h, _Float16* w1h, _Float16* w2h,
    float* bnE, float* bnH, float* w3o)
{
    int tid = threadIdx.x;
    for (int i = tid; i < 64 * 128;  i += 256) wk1h[i] = (_Float16)wk1f[i];
    for (int i = tid; i < 64 * 64;   i += 256) wk2h[i] = (_Float16)wk2f[i];
    for (int i = tid; i < 256 * 64;  i += 256) w1h[i]  = (_Float16)w1f[i];
    for (int i = tid; i < 128 * 256; i += 256) w2h[i]  = (_Float16)w2f[i];
    if (tid < 64) {
        float s1 = gk1[tid] * rsqrtf(vk1[tid] + 1e-5f);
        bnE[tid]       = s1;
        bnE[64 + tid]  = bk1[tid] - mk1[tid] * s1;
        float s2 = gk2[tid] * rsqrtf(vk2[tid] + 1e-5f);
        bnE[128 + tid] = s2;
        bnE[192 + tid] = bk2[tid] - mk2[tid] * s2;
    }
    if (tid < 256) {
        float s = g1[tid] * rsqrtf(v1[tid] + 1e-5f);
        bnH[tid]        = s;
        bnH[256 + tid]  = b1[tid] - m1[tid] * s;
    }
    if (tid < 128) {
        float s = g2[tid] * rsqrtf(v2[tid] + 1e-5f);
        bnH[512 + tid]  = s;
        bnH[640 + tid]  = b2[tid] - m2[tid] * s;
        w3o[tid] = w3[tid];
    }
    if (tid == 0) w3o[128] = b3[0];
}

// ---------------------------------------------------------------------------
// x (B,C,N) -> xt (B,N,C): coalesced reads over n, row-contiguous points.
// ---------------------------------------------------------------------------
__global__ __launch_bounds__(256) void transpose_x(const float* __restrict__ x,
                                                   float* __restrict__ xt)
{
    int g = blockIdx.x * 256 + threadIdx.x;    // over B*N
    if (g >= BATCH * NPTS) return;
    int b = g >> 12, n = g & (NPTS - 1);
    const float* xb = x + (size_t)b * CH * NPTS;
    float* o = xt + (size_t)g * CH;
    #pragma unroll 8
    for (int c = 0; c < CH; ++c) o[c] = xb[c * NPTS + n];
}

// ---------------------------------------------------------------------------
// kNN: Gram tiles via V_WMMA_F32_16X16X4_F32, streaming register top-20.
// Column tiles staged with GLOBAL_LOAD_ASYNC_TO_LDS_B128 into padded LDS.
// Block = 256 threads (8 waves), owns 128 rows vs all 4096 columns.
// ---------------------------------------------------------------------------
#define TS 128
#define ROWPAD 68   // 64 + 4 floats: avoids 16-way bank conflict on frag reads
__global__ __launch_bounds__(256) void knn_kernel(const float* __restrict__ xt,
                                                  int* __restrict__ knnIdx)
{
    __shared__ __align__(16) float rowsL[TS][ROWPAD];
    __shared__ __align__(16) float colsL[TS][ROWPAD];
    __shared__ float xxr[TS];
    __shared__ float xxc[TS];
    __shared__ __align__(16) float dtile[TS][TS + 4];

    int b = blockIdx.y;
    int rowBase = blockIdx.x * TS;
    int tid = threadIdx.x, lane = tid & 31, wave = tid >> 5;
    const float* xb = xt + (size_t)b * NPTS * CH;

    unsigned rowsLds = (unsigned)(uintptr_t)(void*)rowsL;
    unsigned colsLds = (unsigned)(uintptr_t)(void*)colsL;

    // Async-stage the 128 x 64 row tile (padded rows -> per-row chunks).
    {
        const float* gsrc = xb + (size_t)rowBase * CH;   // contiguous 32 KB
        for (int i = tid; i < (TS * CH) / 4; i += 256) { // 2048 x 16B
            int r = i >> 4, cw = i & 15;
            async_b128(rowsLds + (unsigned)(r * ROWPAD * 4 + cw * 16),
                       (unsigned)((r * CH + cw * 4) * 4), gsrc);
        }
        async_wait0();
    }
    __syncthreads();
    if (tid < TS) {
        float s = 0.0f;
        for (int c = 0; c < CH; ++c) { float v = rowsL[tid][c]; s += v * v; }
        xxr[tid] = s;
    }

    float best[KNN]; int bidx[KNN];
    #pragma unroll
    for (int i = 0; i < KNN; ++i) { best[i] = -3.0e38f; bidx[i] = 0; }

    int m  = lane & 15;
    int kh = (lane < 16) ? 0 : 2;   // f32 16x4 A/B frag: K pair select per half-wave
    int mo = (lane < 16) ? 0 : 8;   // f32 C/D frag: row offset per half-wave

    for (int ct = 0; ct < NPTS / TS; ++ct) {
        int colBase = ct * TS;
        __syncthreads();
        {
            const float* gsrc = xb + (size_t)colBase * CH;
            for (int i = tid; i < (TS * CH) / 4; i += 256) {
                int r = i >> 4, cw = i & 15;
                async_b128(colsLds + (unsigned)(r * ROWPAD * 4 + cw * 16),
                           (unsigned)((r * CH + cw * 4) * 4), gsrc);
            }
            async_wait0();
        }
        __syncthreads();
        if (tid < TS) {
            float s = 0.0f;
            for (int c = 0; c < CH; ++c) { float v = colsL[tid][c]; s += v * v; }
            xxc[tid] = s;
        }
        __syncthreads();

        // 8x8 grid of 16x16 tiles; each wave takes 8 tiles.
        for (int t = wave; t < 64; t += 8) {
            int tr = (t >> 3) * 16, tc = (t & 7) * 16;
            v8f acc = {};
            #pragma unroll
            for (int kk = 0; kk < CH; kk += 4) {
                v2f av, bv;
                av[0] = rowsL[tr + m][kk + kh];
                av[1] = rowsL[tr + m][kk + kh + 1];
                bv[0] = colsL[tc + m][kk + kh];
                bv[1] = colsL[tc + m][kk + kh + 1];
                acc = __builtin_amdgcn_wmma_f32_16x16x4_f32(
                        false, av, false, bv, (short)0, acc, false, false);
            }
            #pragma unroll
            for (int e = 0; e < 8; ++e) {
                int rr = tr + mo + e;
                dtile[rr][tc + m] = 2.0f * acc[e] - xxr[rr] - xxc[tc + m];
            }
        }
        __syncthreads();

        if (tid < TS) {
            #pragma unroll 4
            for (int j = 0; j < TS; ++j) {
                float d = dtile[tid][j];
                if (d > best[KNN - 1]) {
                    float cd = d; int ci = colBase + j;
                    #pragma unroll
                    for (int i = 0; i < KNN; ++i) {
                        if (cd > best[i]) {
                            float tf = best[i]; int ti = bidx[i];
                            best[i] = cd; bidx[i] = ci;
                            cd = tf; ci = ti;
                        }
                    }
                }
            }
        }
    }
    if (tid < TS) {
        int* o = knnIdx + ((size_t)b * NPTS + rowBase + tid) * KNN;
        #pragma unroll
        for (int i = 0; i < KNN; ++i) o[i] = bidx[i];
    }
}

// ---------------------------------------------------------------------------
// Edge convs: feat(128 x 320) -> conv1(64) -> conv2(64) -> max over k.
// 16 points/block, WMMA f32_16x16x32_f16; weights staged via async copy.
// ---------------------------------------------------------------------------
#define PB 16
#define CB (PB * KNN)   // 320 columns
__global__ __launch_bounds__(256) void edge_kernel(
    const float* __restrict__ xt, const int* __restrict__ knnIdx,
    const _Float16* __restrict__ wk1h, const _Float16* __restrict__ wk2h,
    const float* __restrict__ bnE, _Float16* __restrict__ hbuf)
{
    __shared__ __align__(16) _Float16 featL[CB * 128];   // 80 KB, col-major [col][ch]
    __shared__ __align__(16) _Float16 f1L[CB * 64];      // 40 KB
    __shared__ __align__(16) _Float16 w1L[64 * 128];     // 16 KB
    __shared__ __align__(16) _Float16 w2L[64 * 64];      //  8 KB
    __shared__ __align__(16) float cenL[PB * 64];        //  4 KB
    __shared__ float sc1[64], bi1[64], sc2[64], bi2[64];
    float* f2L = (float*)featL;                          // reuse after conv1 (80 KB)

    int b = blockIdx.y;
    int p0 = blockIdx.x * PB;
    int tid = threadIdx.x, lane = tid & 31, wave = tid >> 5;

    // Async-stage weights and this block's center features.
    async_copy_lin(w1L, wk1h, 64 * 128 * 2, tid);
    async_copy_lin(w2L, wk2h, 64 * 64 * 2, tid);
    async_copy_lin(cenL, xt + ((size_t)b * NPTS + p0) * CH, PB * 64 * 4, tid);
    if (tid < 64) {
        sc1[tid] = bnE[tid];       bi1[tid] = bnE[64 + tid];
        sc2[tid] = bnE[128 + tid]; bi2[tid] = bnE[192 + tid];
    }
    async_wait0();
    __syncthreads();

    // Gather feature columns: [neigh - center ; center]
    for (int col = tid; col < CB; col += 256) {
        int p = col / KNN, j = col - p * KNN;
        int nb = knnIdx[((size_t)b * NPTS + p0 + p) * KNN + j];
        const float4* xn = (const float4*)(xt + ((size_t)b * NPTS + nb) * CH);
        const float4* cn = (const float4*)(cenL + p * 64);
        _Float16* fc = featL + col * 128;
        #pragma unroll
        for (int q = 0; q < 16; ++q) {
            float4 nv = xn[q], cv = cn[q];
            int cc = q * 4;
            fc[cc + 0]      = (_Float16)(nv.x - cv.x);
            fc[cc + 1]      = (_Float16)(nv.y - cv.y);
            fc[cc + 2]      = (_Float16)(nv.z - cv.z);
            fc[cc + 3]      = (_Float16)(nv.w - cv.w);
            fc[64 + cc + 0] = (_Float16)cv.x;
            fc[64 + cc + 1] = (_Float16)cv.y;
            fc[64 + cc + 2] = (_Float16)cv.z;
            fc[64 + cc + 3] = (_Float16)cv.w;
        }
    }
    __syncthreads();

    int m    = lane & 15;
    int nco  = lane & 15;
    int koA  = (lane < 16) ? 0 : 16;  // f16 A-frag K sub-offset
    int kbB  = (lane < 16) ? 0 : 16;  // f16 B-frag K base per half-wave
    int mo   = (lane < 16) ? 0 : 8;

    // Conv1: 64x128 weights, 320 columns -> 4 x 20 tiles, 4 K-steps
    for (int t = wave; t < 80; t += 8) {
        int mt = t & 3, nt = t >> 2;
        v8f acc = {};
        #pragma unroll
        for (int kb = 0; kb < 128; kb += 32) {
            v16h a, bf;
            const _Float16* wr = w1L + (mt * 16 + m) * 128 + kb + koA;
            #pragma unroll
            for (int h = 0; h < 8; ++h) { a[h] = wr[h]; a[h + 8] = wr[8 + h]; }
            const _Float16* bp = featL + (nt * 16 + nco) * 128 + kb + kbB;
            #pragma unroll
            for (int h = 0; h < 16; ++h) bf[h] = bp[h];
            acc = __builtin_amdgcn_wmma_f32_16x16x32_f16(
                    false, a, false, bf, (short)0, acc, false, false);
        }
        #pragma unroll
        for (int e = 0; e < 8; ++e) {
            int ch = mt * 16 + mo + e;
            float v = lrelu(acc[e] * sc1[ch] + bi1[ch]);
            f1L[(nt * 16 + nco) * 64 + ch] = (_Float16)v;
        }
    }
    __syncthreads();

    // Conv2: 64x64 weights, 2 K-steps; results (f32) overwrite featL region
    for (int t = wave; t < 80; t += 8) {
        int mt = t & 3, nt = t >> 2;
        v8f acc = {};
        #pragma unroll
        for (int kb = 0; kb < 64; kb += 32) {
            v16h a, bf;
            const _Float16* wr = w2L + (mt * 16 + m) * 64 + kb + koA;
            #pragma unroll
            for (int h = 0; h < 8; ++h) { a[h] = wr[h]; a[h + 8] = wr[8 + h]; }
            const _Float16* bp = f1L + (nt * 16 + nco) * 64 + kb + kbB;
            #pragma unroll
            for (int h = 0; h < 16; ++h) bf[h] = bp[h];
            acc = __builtin_amdgcn_wmma_f32_16x16x32_f16(
                    false, a, false, bf, (short)0, acc, false, false);
        }
        #pragma unroll
        for (int e = 0; e < 8; ++e) {
            int ch = mt * 16 + mo + e;
            float v = lrelu(acc[e] * sc2[ch] + bi2[ch]);
            f2L[(nt * 16 + nco) * 64 + ch] = v;
        }
    }
    __syncthreads();

    // Max over k per (point, channel)
    for (int i = tid; i < PB * 64; i += 256) {
        int p = i >> 6, c = i & 63;
        float mx = -3.0e38f;
        #pragma unroll 5
        for (int j = 0; j < KNN; ++j) mx = fmaxf(mx, f2L[(p * KNN + j) * 64 + c]);
        hbuf[((size_t)b * NPTS + p0 + p) * 64 + c] = (_Float16)mx;
    }
}

// ---------------------------------------------------------------------------
// Head MLP: h(64) -> 256 -> 128 -> 1.  64 points/block.
// ---------------------------------------------------------------------------
#define PC 64
__global__ __launch_bounds__(256) void head_kernel(
    const _Float16* __restrict__ hbuf,
    const _Float16* __restrict__ w1h, const _Float16* __restrict__ w2h,
    const float* __restrict__ bnH, const float* __restrict__ w3o,
    float* __restrict__ out)
{
    __shared__ __align__(16) _Float16 hL[PC * 64];       //  8 KB
    __shared__ __align__(16) _Float16 w1Ls[256 * 64];    // 32 KB
    __shared__ __align__(16) _Float16 g1L[PC * 256];     // 32 KB
    __shared__ __align__(16) _Float16 w2Ls[128 * 256];   // 64 KB
    __shared__ __align__(16) float    g2L[PC * 128];     // 32 KB
    __shared__ float scA[256], biA[256], scB[128], biB[128], w3s[129];

    int b = blockIdx.y;
    int p0 = blockIdx.x * PC;
    int tid = threadIdx.x, lane = tid & 31, wave = tid >> 5;

    async_copy_lin(hL,   hbuf + ((size_t)b * NPTS + p0) * 64, PC * 64 * 2, tid);
    async_copy_lin(w1Ls, w1h, 256 * 64 * 2, tid);
    async_copy_lin(w2Ls, w2h, 128 * 256 * 2, tid);
    if (tid < 256) { scA[tid] = bnH[tid]; biA[tid] = bnH[256 + tid]; }
    if (tid < 128) { scB[tid] = bnH[512 + tid]; biB[tid] = bnH[640 + tid]; w3s[tid] = w3o[tid]; }
    if (tid == 0) w3s[128] = w3o[128];
    async_wait0();
    __syncthreads();

    int m   = lane & 15;
    int nco = lane & 15;
    int koA = (lane < 16) ? 0 : 16;
    int kbB = (lane < 16) ? 0 : 16;
    int mo  = (lane < 16) ? 0 : 8;

    // Layer 1: 256x64, 16 mt x 4 nt tiles, 2 K-steps
    for (int t = wave; t < 64; t += 8) {
        int mt = t >> 2, nt = t & 3;
        v8f acc = {};
        #pragma unroll
        for (int kb = 0; kb < 64; kb += 32) {
            v16h a, bf;
            const _Float16* wr = w1Ls + (mt * 16 + m) * 64 + kb + koA;
            #pragma unroll
            for (int h = 0; h < 8; ++h) { a[h] = wr[h]; a[h + 8] = wr[8 + h]; }
            const _Float16* bp = hL + (nt * 16 + nco) * 64 + kb + kbB;
            #pragma unroll
            for (int h = 0; h < 16; ++h) bf[h] = bp[h];
            acc = __builtin_amdgcn_wmma_f32_16x16x32_f16(
                    false, a, false, bf, (short)0, acc, false, false);
        }
        #pragma unroll
        for (int e = 0; e < 8; ++e) {
            int ch = mt * 16 + mo + e;
            float v = lrelu(acc[e] * scA[ch] + biA[ch]);
            g1L[(nt * 16 + nco) * 256 + ch] = (_Float16)v;
        }
    }
    __syncthreads();

    // Layer 2: 128x256, 8 mt x 4 nt tiles, 8 K-steps
    for (int t = wave; t < 32; t += 8) {
        int mt = t >> 2, nt = t & 3;
        v8f acc = {};
        #pragma unroll
        for (int kb = 0; kb < 256; kb += 32) {
            v16h a, bf;
            const _Float16* wr = w2Ls + (mt * 16 + m) * 256 + kb + koA;
            #pragma unroll
            for (int h = 0; h < 8; ++h) { a[h] = wr[h]; a[h + 8] = wr[8 + h]; }
            const _Float16* bp = g1L + (nt * 16 + nco) * 256 + kb + kbB;
            #pragma unroll
            for (int h = 0; h < 16; ++h) bf[h] = bp[h];
            acc = __builtin_amdgcn_wmma_f32_16x16x32_f16(
                    false, a, false, bf, (short)0, acc, false, false);
        }
        #pragma unroll
        for (int e = 0; e < 8; ++e) {
            int ch = mt * 16 + mo + e;
            float v = lrelu(acc[e] * scB[ch] + biB[ch]);
            g2L[(nt * 16 + nco) * 128 + ch] = v;
        }
    }
    __syncthreads();

    // Final 1x128 projection
    if (tid < PC) {
        float s = w3s[128];
        #pragma unroll 8
        for (int c = 0; c < 128; ++c) s += w3s[c] * g2L[tid * 128 + c];
        out[(size_t)b * NPTS + p0 + tid] = s;
    }
}

// ---------------------------------------------------------------------------
extern "C" void kernel_launch(void* const* d_in, const int* in_sizes, int n_in,
                              void* d_out, int out_size, void* d_ws, size_t ws_size,
                              hipStream_t stream) {
    const float* x    = (const float*)d_in[0];
    const float* wk1  = (const float*)d_in[1];
    const float* gk1  = (const float*)d_in[2];
    const float* bk1  = (const float*)d_in[3];
    const float* mk1  = (const float*)d_in[4];
    const float* vk1  = (const float*)d_in[5];
    const float* wk2  = (const float*)d_in[6];
    const float* gk2  = (const float*)d_in[7];
    const float* bk2  = (const float*)d_in[8];
    const float* mk2  = (const float*)d_in[9];
    const float* vk2  = (const float*)d_in[10];
    const float* w1   = (const float*)d_in[11];
    const float* g1   = (const float*)d_in[12];
    const float* b1   = (const float*)d_in[13];
    const float* m1   = (const float*)d_in[14];
    const float* v1   = (const float*)d_in[15];
    const float* w2   = (const float*)d_in[16];
    const float* g2   = (const float*)d_in[17];
    const float* b2   = (const float*)d_in[18];
    const float* m2   = (const float*)d_in[19];
    const float* v2   = (const float*)d_in[20];
    const float* w3   = (const float*)d_in[21];
    const float* b3   = (const float*)d_in[22];
    float* out = (float*)d_out;

    char* ws = (char*)d_ws;
    size_t off = 0;
    int*      knnIdx = (int*)(ws + off);      off += (size_t)BATCH * NPTS * KNN * 4;   // 2.62 MB
    float*    xt     = (float*)(ws + off);    off += (size_t)BATCH * NPTS * CH * 4;    // 8.39 MB
    _Float16* hbuf   = (_Float16*)(ws + off); off += (size_t)BATCH * NPTS * CH * 2;    // 4.19 MB
    _Float16* wk1h   = (_Float16*)(ws + off); off += 64 * 128 * 2;
    _Float16* wk2h   = (_Float16*)(ws + off); off += 64 * 64 * 2;
    _Float16* w1h    = (_Float16*)(ws + off); off += 256 * 64 * 2;
    _Float16* w2h    = (_Float16*)(ws + off); off += 128 * 256 * 2;
    float*    bnE    = (float*)(ws + off);    off += 256 * 4;
    float*    bnH    = (float*)(ws + off);    off += 768 * 4;
    float*    w3o    = (float*)(ws + off);    off += 129 * 4;

    prep_params<<<1, 256, 0, stream>>>(
        wk1, gk1, bk1, mk1, vk1, wk2, gk2, bk2, mk2, vk2,
        w1, g1, b1, m1, v1, w2, g2, b2, m2, v2, w3, b3,
        wk1h, wk2h, w1h, w2h, bnE, bnH, w3o);

    transpose_x<<<(BATCH * NPTS + 255) / 256, 256, 0, stream>>>(x, xt);

    knn_kernel<<<dim3(NPTS / TS, BATCH), 256, 0, stream>>>(xt, knnIdx);

    edge_kernel<<<dim3(NPTS / PB, BATCH), 256, 0, stream>>>(
        xt, knnIdx, wk1h, wk2h, bnE, hbuf);

    head_kernel<<<dim3(NPTS / PC, BATCH), 256, 0, stream>>>(
        hbuf, w1h, w2h, bnH, w3o, out);
}
